// MatchModule_50053548868047
// MI455X (gfx1250) — compile-verified, matrix operand order checked by reading, not run on hardware
//
#include <hip/hip_runtime.h>
#include <hip/hip_bf16.h>

typedef __attribute__((ext_vector_type(16))) _Float16 v16h;
typedef __attribute__((ext_vector_type(8)))  _Float16 v8h;
typedef __attribute__((ext_vector_type(8)))  float    v8f;
typedef __attribute__((ext_vector_type(4)))  float    v4f;

static __device__ __forceinline__ v8f wmma16(v16h a, v16h b, v8f c) {
  // D = A(16x32 f16) * B(32x16 f16) + C(16x16 f32)
  return __builtin_amdgcn_wmma_f32_16x16x32_f16(false, a, false, b, (short)0, c, false, false);
}

// K-offset of element e (0..15) in the 16-bit 16x32 A-fragment layout:
// lanes 0-15: K = {0..7, 16..23}; lanes 16-31: K = {8..15, 24..31}
static __device__ __forceinline__ int kofs(int e, int half) {
  return ((e >> 3) << 4) + (half << 3) + (e & 7);
}

// Load a 16x32-f16 fragment row from f32 memory using 4x b128 loads.
// p = row base (already includes the k-chunk offset); lane's two runs are
// [half*8, half*8+8) and [16+half*8, 16+half*8+8).
static __device__ __forceinline__ v16h cvt_frag(const float* __restrict__ p, int half) {
  const v4f* q0 = (const v4f*)(p + half * 8);
  const v4f* q1 = (const v4f*)(p + 16 + half * 8);
  v4f f0 = q0[0], f1 = q0[1], f2 = q1[0], f3 = q1[1];
  v16h a;
#pragma unroll
  for (int j = 0; j < 4; ++j) {
    a[j]      = (_Float16)f0[j];
    a[4 + j]  = (_Float16)f1[j];
    a[8 + j]  = (_Float16)f2[j];
    a[12 + j] = (_Float16)f3[j];
  }
  return a;
}

// ---------------------------------------------------------------------------
// distance bias: bias[b,0,q,k] = w(q,k)/sum_q' w(q',k), bias[b,1,q,k] = -dist
// ---------------------------------------------------------------------------
__global__ void dist_bias_kernel(const float* __restrict__ center,
                                 float* __restrict__ bias, int Bn, int Kn) {
  int idx = blockIdx.x * blockDim.x + threadIdx.x;
  if (idx >= Bn * Kn) return;
  int b = idx / Kn, kc = idx % Kn;
  const float* cb = center + (size_t)b * Kn * 3;
  float xk = cb[kc * 3 + 0], yk = cb[kc * 3 + 1], zk = cb[kc * 3 + 2];
  float s = 0.f;
  for (int q = 0; q < Kn; ++q) {
    float dx = cb[q * 3 + 0] - xk, dy = cb[q * 3 + 1] - yk, dz = cb[q * 3 + 2] - zk;
    float d = sqrtf(dx * dx + dy * dy + dz * dz);
    s += 1.f / (d + 0.01f);
  }
  float inv = 1.f / s;
  float* b0 = bias + ((size_t)b * 2 + 0) * Kn * Kn;
  float* b1 = bias + ((size_t)b * 2 + 1) * Kn * Kn;
  for (int q = 0; q < Kn; ++q) {
    float dx = cb[q * 3 + 0] - xk, dy = cb[q * 3 + 1] - yk, dz = cb[q * 3 + 2] - zk;
    float d = sqrtf(dx * dx + dy * dy + dz * dz);
    b0[(size_t)q * Kn + kc] = (1.f / (d + 0.01f)) * inv;
    b1[(size_t)q * Kn + kc] = -d;
  }
}

// ---------------------------------------------------------------------------
// Stage a 32x128 f32 weight slab into LDS pre-swizzled into B-fragment order:
// wlds[(nt*32 + lane)*16 + e] = (f16) W[(kc+k)*128 + n]
// so each lane later reads its whole fragment as one aligned 32B LDS access.
// ---------------------------------------------------------------------------
static __device__ __forceinline__ void stage_wslab(const float* __restrict__ W,
                                                   int kc, _Float16* wlds, int tid) {
  for (int i = tid; i < 32 * 128; i += 128) {
    int k = i >> 7, n = i & 127;  // coalesced global read across n for fixed k
    float w = W[(size_t)(kc + k) * 128 + n];
    int e = ((k >> 4) << 3) | (k & 7);
    int lane_d = (n & 15) | (((k >> 3) & 1) << 4);
    int nt = n >> 4;
    wlds[((nt * 32 + lane_d) << 4) | e] = (_Float16)w;
  }
}

// ---------------------------------------------------------------------------
// C[rows,128] = act(A[rows,Kdim] @ W[Kdim,128] + bias)
// bn != nullptr: BatchNorm (running stats) + PReLU epilogue
// 4 waves / block; wave owns 16 rows x 128 cols (8 WMMA accumulator tiles)
// ---------------------------------------------------------------------------
__global__ __launch_bounds__(128) void gemm128_kernel(
    const float* __restrict__ A, const float* __restrict__ W,
    const float* __restrict__ bias, float* __restrict__ C,
    int rows, int Kdim, const float* __restrict__ bn,
    const float* __restrict__ alpha, int alpha_stride) {
  __shared__ _Float16 wlds[8 * 32 * 16];
  const int tid = threadIdx.x;
  const int wave = tid >> 5, lane = tid & 31;
  const int half = lane >> 4, nn = lane & 15;
  const int rowbase = blockIdx.x * 64 + wave * 16;
  (void)rows;

  v8f c[8];
#pragma unroll
  for (int nt = 0; nt < 8; ++nt) c[nt] = (v8f){0.f, 0.f, 0.f, 0.f, 0.f, 0.f, 0.f, 0.f};

  for (int kc = 0; kc < Kdim; kc += 32) {
    __syncthreads();
    stage_wslab(W, kc, wlds, tid);
    __syncthreads();
    const float* ap = A + (size_t)(rowbase + nn) * Kdim + kc;
    if (kc + 32 < Kdim) __builtin_prefetch(ap + 32, 0, 1);  // global_prefetch
    v16h a = cvt_frag(ap, half);
#pragma unroll
    for (int nt = 0; nt < 8; ++nt) {
      v16h b = *(const v16h*)(wlds + ((nt * 32 + lane) << 4));
      c[nt] = wmma16(a, b, c[nt]);
    }
  }

#pragma unroll
  for (int nt = 0; nt < 8; ++nt) {
#pragma unroll
    for (int v = 0; v < 8; ++v) {
      int m = v + 8 * half;
      int n = nt * 16 + nn;
      float x = c[nt][v] + bias[n];
      if (bn) {
        float g = bn[n], bb = bn[128 + n], mm = bn[256 + n], vv = bn[384 + n];
        x = g * (x - mm) * rsqrtf(vv + 1e-5f) + bb;
        float al = alpha[alpha_stride * n];
        x = (x >= 0.f) ? x : al * x;
      }
      C[(size_t)(rowbase + m) * 128 + n] = x;
    }
  }
}

// ---------------------------------------------------------------------------
// C = LayerNorm(resid + ATT @ Wo + bo) ; gamma/beta per channel
// ---------------------------------------------------------------------------
__global__ __launch_bounds__(128) void wo_ln_kernel(
    const float* __restrict__ ATT, const float* __restrict__ Wo,
    const float* __restrict__ bo, const float* __restrict__ resid,
    const float* __restrict__ gamma, const float* __restrict__ beta,
    float* __restrict__ C, int rows) {
  __shared__ _Float16 wlds[8 * 32 * 16];
  const int tid = threadIdx.x;
  const int wave = tid >> 5, lane = tid & 31;
  const int half = lane >> 4, nn = lane & 15;
  const int rowbase = blockIdx.x * 64 + wave * 16;
  (void)rows;

  v8f c[8];
#pragma unroll
  for (int nt = 0; nt < 8; ++nt) c[nt] = (v8f){0.f, 0.f, 0.f, 0.f, 0.f, 0.f, 0.f, 0.f};

  for (int kc = 0; kc < 128; kc += 32) {
    __syncthreads();
    stage_wslab(Wo, kc, wlds, tid);
    __syncthreads();
    v16h a = cvt_frag(ATT + (size_t)(rowbase + nn) * 128 + kc, half);
#pragma unroll
    for (int nt = 0; nt < 8; ++nt) {
      v16h b = *(const v16h*)(wlds + ((nt * 32 + lane) << 4));
      c[nt] = wmma16(a, b, c[nt]);
    }
  }

  float s1[8], s2[8];
#pragma unroll
  for (int v = 0; v < 8; ++v) { s1[v] = 0.f; s2[v] = 0.f; }
#pragma unroll
  for (int nt = 0; nt < 8; ++nt) {
#pragma unroll
    for (int v = 0; v < 8; ++v) {
      int m = v + 8 * half;
      int n = nt * 16 + nn;
      float x = c[nt][v] + bo[n] + resid[(size_t)(rowbase + m) * 128 + n];
      c[nt][v] = x;
      s1[v] += x;
      s2[v] += x * x;
    }
  }
#pragma unroll
  for (int v = 0; v < 8; ++v) {
#pragma unroll
    for (int off = 8; off >= 1; off >>= 1) {
      s1[v] += __shfl_xor(s1[v], off, 32);
      s2[v] += __shfl_xor(s2[v], off, 32);
    }
  }
#pragma unroll
  for (int nt = 0; nt < 8; ++nt) {
#pragma unroll
    for (int v = 0; v < 8; ++v) {
      int m = v + 8 * half;
      int n = nt * 16 + nn;
      float mean = s1[v] * (1.f / 128.f);
      float var = s2[v] * (1.f / 128.f) - mean * mean;
      float iv = rsqrtf(var + 1e-5f);
      C[(size_t)(rowbase + m) * 128 + n] = gamma[n] * (c[nt][v] - mean) * iv + beta[n];
    }
  }
}

// ---------------------------------------------------------------------------
// Fused attention, one block per (q-block of 64, head, batch); 4 waves.
// NKT = number of 16-wide key tiles (nk = NKT*16). dk = 32 (one WMMA K-step).
// ---------------------------------------------------------------------------
template <int NKT>
__global__ __launch_bounds__(128) void attn_kernel(
    const float* __restrict__ Qb, const float* __restrict__ Kb,
    const float* __restrict__ Vb, const float* __restrict__ bias_base,
    int bias_div, const unsigned char* __restrict__ maskp,
    float* __restrict__ Ob, int nq, int nk) {
  constexpr int KP = ((NKT * 16 + 31) / 32) * 32;  // keys padded to WMMA K-step
  __shared__ _Float16 Psh[4 * 16 * KP];
  const int tid = threadIdx.x;
  const int wave = tid >> 5, lane = tid & 31;
  const int half = lane >> 4, nn = lane & 15;
  const int bz = blockIdx.z, h = blockIdx.y;
  const int qbase = blockIdx.x * 64 + wave * 16;
  const float scale = 0.176776695296636881f;  // 1/sqrt(32)

  // Q fragment (16 rows x 32 dk) via b128 loads
  v16h aq = cvt_frag(Qb + ((size_t)bz * nq + qbase + nn) * 128 + h * 32, half);

  // scores S = Q K^T : NKT tiles of 16x16 kept in registers
  v8f cs[NKT];
#pragma unroll
  for (int kt = 0; kt < NKT; ++kt) {
    v16h bk = cvt_frag(Kb + ((size_t)bz * nk + kt * 16 + nn) * 128 + h * 32, half);
    v8f z = (v8f){0.f, 0.f, 0.f, 0.f, 0.f, 0.f, 0.f, 0.f};
    cs[kt] = wmma16(aq, bk, z);
  }

  const float* bp = nullptr;
  if (bias_base && h < 2)
    bp = bias_base + ((size_t)(bz / bias_div) * 2 + h) * (size_t)nq * nk;
  const unsigned char* mp = maskp ? (maskp + (size_t)bz * nk) : nullptr;

  // softmax over all keys: scale + bias + mask, row max, exp, row sum
  float rm[8];
#pragma unroll
  for (int v = 0; v < 8; ++v) rm[v] = -3e38f;
#pragma unroll
  for (int kt = 0; kt < NKT; ++kt) {
#pragma unroll
    for (int v = 0; v < 8; ++v) {
      int q = qbase + v + 8 * half;
      int k = kt * 16 + nn;
      float s = cs[kt][v] * scale;
      if (bp) s += bp[(size_t)q * nk + k];
      if (mp && mp[k]) s = -3e38f;
      cs[kt][v] = s;
      rm[v] = fmaxf(rm[v], s);
    }
  }
#pragma unroll
  for (int v = 0; v < 8; ++v)
#pragma unroll
    for (int off = 8; off >= 1; off >>= 1) rm[v] = fmaxf(rm[v], __shfl_xor(rm[v], off, 32));

  float rs[8];
#pragma unroll
  for (int v = 0; v < 8; ++v) rs[v] = 0.f;
#pragma unroll
  for (int kt = 0; kt < NKT; ++kt) {
#pragma unroll
    for (int v = 0; v < 8; ++v) {
      float p = __expf(cs[kt][v] - rm[v]);
      cs[kt][v] = p;
      rs[v] += p;
    }
  }
#pragma unroll
  for (int v = 0; v < 8; ++v)
#pragma unroll
    for (int off = 8; off >= 1; off >>= 1) rs[v] += __shfl_xor(rs[v], off, 32);

  // write normalized probabilities as f16 to per-wave LDS region (A-readable)
  _Float16* P = Psh + wave * 16 * KP;
#pragma unroll
  for (int kt = 0; kt < NKT; ++kt) {
#pragma unroll
    for (int v = 0; v < 8; ++v) {
      int m = v + 8 * half;
      P[m * KP + kt * 16 + nn] = (_Float16)(cs[kt][v] * (1.f / rs[v]));
    }
  }
  if constexpr (NKT * 16 < KP) {
    if (lane < 16)
      for (int cpad = NKT * 16; cpad < KP; ++cpad) P[lane * KP + cpad] = (_Float16)0.f;
  }
  // same-wave LDS ops are in-order; no barrier needed

  // O = P @ V : out 16x32 (two n-halves)
  v8f o0 = (v8f){0.f, 0.f, 0.f, 0.f, 0.f, 0.f, 0.f, 0.f};
  v8f o1 = (v8f){0.f, 0.f, 0.f, 0.f, 0.f, 0.f, 0.f, 0.f};
#pragma unroll
  for (int kc = 0; kc < KP / 32; ++kc) {
    const _Float16* prow = P + nn * KP + kc * 32 + half * 8;
    v8h lo = *(const v8h*)prow;
    v8h hi = *(const v8h*)(prow + 16);
    v16h ap = __builtin_shufflevector(lo, hi, 0, 1, 2, 3, 4, 5, 6, 7,
                                      8, 9, 10, 11, 12, 13, 14, 15);
    v16h b0, b1;
#pragma unroll
    for (int e = 0; e < 16; ++e) {
      int key = kc * 32 + kofs(e, half);
      float v0 = 0.f, v1 = 0.f;
      if (KP == NKT * 16 || key < nk) {  // constexpr-pruned for full tiles
        const float* vp = Vb + ((size_t)bz * nk + key) * 128 + h * 32;
        v0 = vp[nn];
        v1 = vp[16 + nn];
      }
      b0[e] = (_Float16)v0;
      b1[e] = (_Float16)v1;
    }
    o0 = wmma16(ap, b0, o0);
    o1 = wmma16(ap, b1, o1);
  }

#pragma unroll
  for (int v = 0; v < 8; ++v) {
    int qrow = qbase + v + 8 * half;
    float* op = Ob + ((size_t)bz * nq + qrow) * 128 + h * 32;
    op[nn] = o0[v];
    op[16 + nn] = o1[v];
  }
}

// ---------------------------------------------------------------------------
__global__ void broadcast_kernel(const float* __restrict__ in, float* __restrict__ out,
                                 int total) {
  int idx = blockIdx.x * blockDim.x + threadIdx.x;
  if (idx >= total) return;
  int d = idx & 127;
  int row = idx >> 7;      // (b*M + mi)*256 + q
  int q = row & 255;
  int bm = row >> 8;
  int b = bm >> 3;         // M = 8
  out[idx] = in[((size_t)b * 256 + q) * 128 + d];
}

__global__ void head_kernel(const float* __restrict__ y, const float* __restrict__ w3,
                            const float* __restrict__ b3, float* __restrict__ out,
                            int rows) {
  int r = blockIdx.x * blockDim.x + threadIdx.x;
  if (r >= rows) return;
  const v4f* yp = (const v4f*)(y + (size_t)r * 128);
  const v4f* wp = (const v4f*)w3;
  float s = b3[0];
#pragma unroll
  for (int d = 0; d < 32; ++d) {
    v4f a = yp[d], b = wp[d];
    s += a[0] * b[0] + a[1] * b[1] + a[2] * b[2] + a[3] * b[3];
  }
  out[r] = s;
}

// ---------------------------------------------------------------------------
extern "C" void kernel_launch(void* const* d_in, const int* in_sizes, int n_in,
                              void* d_out, int out_size, void* d_ws, size_t ws_size,
                              hipStream_t stream) {
  (void)in_sizes; (void)n_in; (void)out_size; (void)ws_size;
  const float* center      = (const float*)d_in[0];
  const float* detr        = (const float*)d_in[1];
  const float* lang        = (const float*)d_in[2];
  const unsigned char* msk = (const unsigned char*)d_in[3];
  const float* fc1_W   = (const float*)d_in[7];
  const float* fc1_b   = (const float*)d_in[8];
  const float* bn0     = (const float*)d_in[9];
  const float* prelu0  = (const float*)d_in[10];
  const float* fc2_W   = (const float*)d_in[11];
  const float* fc2_b   = (const float*)d_in[12];
  const float* s_qkv   = (const float*)d_in[13];
  const float* s_qkv_b = (const float*)d_in[14];
  const float* s_Wo    = (const float*)d_in[15];
  const float* s_bo    = (const float*)d_in[16];
  const float* s_ln    = (const float*)d_in[17];
  const float* c_qkv   = (const float*)d_in[18];
  const float* c_qkv_b = (const float*)d_in[19];
  const float* c_Wo    = (const float*)d_in[20];
  const float* c_bo    = (const float*)d_in[21];
  const float* c_ln    = (const float*)d_in[22];
  const float* m_W     = (const float*)d_in[23];
  const float* m_b     = (const float*)d_in[24];
  const float* m_bn    = (const float*)d_in[25];
  const float* m_al    = (const float*)d_in[26];
  const float* m_W3    = (const float*)d_in[27];
  const float* m_b3    = (const float*)d_in[28];
  float* out = (float*)d_out;

  const int Bn = 16, Kn = 256, M = 8, L = 80, BM = 128;
  const int rows1 = Bn * Kn;   // 4096
  const int rowsF = BM * Kn;   // 32768
  const int rowsL = BM * L;    // 10240
  const int DD = 128 * 128;    // weight matrix elems

  float* ws = (float*)d_ws;
  const size_t SLOT = 4u * 1024u * 1024u;
  float* BIAS = ws;                       // 2*16*256*256 = 2,097,152 floats
  float* S1 = ws + 2097152;
  float* S2 = S1 + SLOT;
  float* S3 = S2 + SLOT;
  float* S4 = S3 + SLOT;
  float* S5 = S4 + SLOT;
  float* S6 = S5 + SLOT;

  // 1) distance attention bias
  dist_bias_kernel<<<16, 256, 0, stream>>>(center, BIAS, Bn, Kn);

  // 2) fc1 + BN + PReLU ; 3) fc2
  gemm128_kernel<<<rows1 / 64, 128, 0, stream>>>(detr, fc1_W, fc1_b, S1, rows1, 1152,
                                                 bn0, prelu0, 1);
  gemm128_kernel<<<rows1 / 64, 128, 0, stream>>>(S1, fc2_W, fc2_b, S2, rows1, 128,
                                                 nullptr, nullptr, 0);

  // 4) self-MHA layer 0 (batch 16, bias = dist_weights)
  for (int j = 0; j < 3; ++j) {
    float* dst = (j == 0) ? S3 : (j == 1) ? S4 : S5;
    gemm128_kernel<<<rows1 / 64, 128, 0, stream>>>(S2, s_qkv + (0 * 3 + j) * DD,
                                                   s_qkv_b + (0 * 3 + j) * 128, dst,
                                                   rows1, 128, nullptr, nullptr, 0);
  }
  attn_kernel<16><<<dim3(4, 4, 16), 128, 0, stream>>>(S3, S4, S5, BIAS, 1, nullptr,
                                                      S6, 256, 256);
  wo_ln_kernel<<<rows1 / 64, 128, 0, stream>>>(S6, s_Wo + 0, s_bo + 0, S2,
                                               s_ln + 0, s_ln + 128, S1, rows1);

  // 5) broadcast (B,K,D) -> (B*M,K,D)
  broadcast_kernel<<<(rowsF * 128) / 256, 256, 0, stream>>>(S1, S2, rowsF * 128);

  // 6) cross-MHA layer 0
  gemm128_kernel<<<rowsF / 64, 128, 0, stream>>>(S2, c_qkv + 0 * DD, c_qkv_b + 0, S3,
                                                 rowsF, 128, nullptr, nullptr, 0);
  gemm128_kernel<<<rowsL / 64, 128, 0, stream>>>(lang, c_qkv + 1 * DD, c_qkv_b + 128, S4,
                                                 rowsL, 128, nullptr, nullptr, 0);
  gemm128_kernel<<<rowsL / 64, 128, 0, stream>>>(lang, c_qkv + 2 * DD, c_qkv_b + 256, S5,
                                                 rowsL, 128, nullptr, nullptr, 0);
  attn_kernel<5><<<dim3(4, 4, 128), 128, 0, stream>>>(S3, S4, S5, nullptr, 1, msk,
                                                      S6, 256, 80);
  wo_ln_kernel<<<rowsF / 64, 128, 0, stream>>>(S6, c_Wo + 0, c_bo + 0, S2,
                                               c_ln + 0, c_ln + 128, S1, rowsF);

  // 7) self-MHA layer 1 (batch 128, bias = dw_rep)
  for (int j = 0; j < 3; ++j) {
    float* dst = (j == 0) ? S3 : (j == 1) ? S4 : S5;
    gemm128_kernel<<<rowsF / 64, 128, 0, stream>>>(S1, s_qkv + (1 * 3 + j) * DD,
                                                   s_qkv_b + (1 * 3 + j) * 128, dst,
                                                   rowsF, 128, nullptr, nullptr, 0);
  }
  attn_kernel<16><<<dim3(4, 4, 128), 128, 0, stream>>>(S3, S4, S5, BIAS, M, nullptr,
                                                       S6, 256, 256);
  wo_ln_kernel<<<rowsF / 64, 128, 0, stream>>>(S6, s_Wo + DD, s_bo + 128, S1,
                                               s_ln + 256, s_ln + 384, S2, rowsF);

  // 8) cross-MHA layer 1
  gemm128_kernel<<<rowsF / 64, 128, 0, stream>>>(S2, c_qkv + 3 * DD, c_qkv_b + 384, S3,
                                                 rowsF, 128, nullptr, nullptr, 0);
  gemm128_kernel<<<rowsL / 64, 128, 0, stream>>>(lang, c_qkv + 4 * DD, c_qkv_b + 512, S4,
                                                 rowsL, 128, nullptr, nullptr, 0);
  gemm128_kernel<<<rowsL / 64, 128, 0, stream>>>(lang, c_qkv + 5 * DD, c_qkv_b + 640, S5,
                                                 rowsL, 128, nullptr, nullptr, 0);
  attn_kernel<5><<<dim3(4, 4, 128), 128, 0, stream>>>(S3, S4, S5, nullptr, 1, msk,
                                                      S6, 256, 80);
  wo_ln_kernel<<<rowsF / 64, 128, 0, stream>>>(S6, c_Wo + DD, c_bo + 128, S2,
                                               c_ln + 256, c_ln + 384, S1, rowsF);

  // 9) match head
  gemm128_kernel<<<rowsF / 64, 128, 0, stream>>>(S1, m_W + 0, m_b + 0, S2, rowsF, 128,
                                                 m_bn + 0, m_al + 0, 0);
  gemm128_kernel<<<rowsF / 64, 128, 0, stream>>>(S2, m_W + DD, m_b + 128, S3, rowsF, 128,
                                                 m_bn + 512, m_al + 1, 0);
  head_kernel<<<rowsF / 256, 256, 0, stream>>>(S3, m_W3, m_b3, out, rowsF);
}
